// WindowAttention_7550552506772
// MI455X (gfx1250) — matrix operand order repaired and checked by reading, not compiled, not run on hardware
//
#include <hip/hip_runtime.h>
#include <hip/hip_bf16.h>

// MI455X / gfx1250 Swin window-attention forward.
// One workgroup (8 wave32) per window; bf16 WMMA (v_wmma_f32_16x16x32_bf16),
// f32 accumulate. Weights pre-swizzled into WMMA B-fragment order in d_ws.

typedef __attribute__((ext_vector_type(16))) __bf16 bf16x16;
typedef __attribute__((ext_vector_type(8)))  float  f32x8;

#define NSEQ   49
#define CDIM   128
#define NHEAD  4
#define HD     32
#define MPAD   64
#define SCALE  0.17677669529663687f   // 32^-0.5

// ws layout (bytes):
//   [0,      98304)  : qkv weights, 96 blocks x 1KB, WMMA-B-fragment swizzled bf16
//   [98304, 131072)  : proj weights, 32 blocks x 1KB, same swizzle
//   [131072,169488)  : bias[4][49][49] f32 (rpb_table gathered via rel_index)
#define WS_QKV_OFF   0
#define WS_PROJ_OFF  98304
#define WS_BIAS_OFF  131072

union FragB { bf16x16 v; uint4 q[2]; };
union FragC { f32x8  v; float f[8]; };

static __device__ __forceinline__ f32x8 wmma_bf16(bf16x16 a, bf16x16 b, f32x8 c) {
  return __builtin_amdgcn_wmma_f32_16x16x32_bf16(false, a, false, b, (short)0, c,
                                                 false, false);
}

// ---------------------------------------------------------------------------
// Prep: swizzle qkv_w / proj_w into B-fragment order; gather rel-pos bias.
// B-fragment layout per 32x16 (KxN) bf16 block (1KB):
//   lane l (0..31): col = nTile*16 + (l&15); K half = 16*(l>>4)
//   dword v (0..7): packs B[kbase+2v], B[kbase+2v+1] at that col
//   address = block*1024 + l*32 + v*4     (block = nTile*4 + kStep)
// B[k][col] = W[col][k]  (since y = x @ W^T)
// ---------------------------------------------------------------------------
__global__ void swin_prep(const float* __restrict__ qkv_w,
                          const float* __restrict__ proj_w,
                          const float* __restrict__ rpb,
                          const int*   __restrict__ rel_index,
                          __bf16* __restrict__ wq,
                          __bf16* __restrict__ wp,
                          float*  __restrict__ bias) {
  int gid = blockIdx.x * blockDim.x + threadIdx.x;
  if (gid < 24576) {                       // 96 blocks * 256 dwords : qkv
    int blk = gid >> 8, r = gid & 255;
    int lane = r >> 3, v = r & 7;
    int nT = blk >> 2, ks = blk & 3;
    int col = nT * 16 + (lane & 15);
    int k0 = ks * 32 + 16 * (lane >> 4) + 2 * v;
    wq[gid * 2 + 0] = (__bf16)qkv_w[col * CDIM + k0];
    wq[gid * 2 + 1] = (__bf16)qkv_w[col * CDIM + k0 + 1];
  } else if (gid < 32768) {                // 32 blocks * 256 dwords : proj
    int g = gid - 24576;
    int blk = g >> 8, r = g & 255;
    int lane = r >> 3, v = r & 7;
    int nT = blk >> 2, ks = blk & 3;
    int col = nT * 16 + (lane & 15);
    int k0 = ks * 32 + 16 * (lane >> 4) + 2 * v;
    wp[g * 2 + 0] = (__bf16)proj_w[col * CDIM + k0];
    wp[g * 2 + 1] = (__bf16)proj_w[col * CDIM + k0 + 1];
  } else {                                 // bias gather: 4*49*49
    int g = gid - 32768;
    if (g < NHEAD * NSEQ * NSEQ) {
      int h = g / (NSEQ * NSEQ), rc = g % (NSEQ * NSEQ);
      bias[g] = rpb[rel_index[rc] * NHEAD + h];
    }
  }
}

// ---------------------------------------------------------------------------
// Main fused attention kernel. blockIdx.x = window index b (4096).
// ---------------------------------------------------------------------------
__global__ __launch_bounds__(256)
void swin_attn(const float* __restrict__ x,
               const float* __restrict__ mask,
               const float* __restrict__ qkv_b,
               const float* __restrict__ proj_b,
               const __bf16* __restrict__ wq,
               const __bf16* __restrict__ wp,
               const float* __restrict__ bias,
               float* __restrict__ out) {
  // 48KB qkv buffer + 16KB scratch (x staging -> S f32 -> P bf16) = 64KB
  __shared__ __bf16 sQKV[MPAD * 3 * CDIM];
  __shared__ float  sScr[MPAD * MPAD];

  const int b    = blockIdx.x;
  const int tid  = threadIdx.x;
  const int wv   = tid >> 5;        // wave id 0..7
  const int lane = tid & 31;
  const int lr   = lane & 15;       // lane row / col within 16
  const int lh   = lane >> 4;       // lane half

  // ---- stage 0: stage x (49x128 f32) -> bf16 64x128, zero-padded rows ----
  {
    __bf16* sX = (__bf16*)sScr;
    const float4* xb4 = (const float4*)(x + (size_t)b * NSEQ * CDIM);
    for (int i = tid; i < MPAD * CDIM / 4; i += 256) {   // 2048 float4
      int r = i >> 5;
      float4 v = (r < NSEQ) ? xb4[i] : make_float4(0.f, 0.f, 0.f, 0.f);
      union { uint2 u; __bf16 hx[4]; } pk;
      pk.hx[0] = (__bf16)v.x; pk.hx[1] = (__bf16)v.y;
      pk.hx[2] = (__bf16)v.z; pk.hx[3] = (__bf16)v.w;
      *(uint2*)(sX + i * 4) = pk.u;
    }
  }
  __syncthreads();

  // ---- stage 1: QKV = X(64x128) @ Wqkv^T(128x384)  : 96 tiles, 12/wave ----
  {
    const __bf16* sX = (const __bf16*)sScr;
    for (int t = wv * 12; t < wv * 12 + 12; ++t) {
      int mT = t & 3, nT = t >> 2;
      FragC acc; acc.v = (f32x8){0.f,0.f,0.f,0.f,0.f,0.f,0.f,0.f};
      for (int ks = 0; ks < 4; ++ks) {
        FragB fa, fb;
        const __bf16* arow = sX + (mT * 16 + lr) * CDIM + ks * 32 + 16 * lh;
        fa.q[0] = *(const uint4*)(arow);
        fa.q[1] = *(const uint4*)(arow + 8);
        const uint4* bp = (const uint4*)(wq + (nT * 4 + ks) * 512 + lane * 16);
        fb.q[0] = bp[0];
        fb.q[1] = bp[1];
        acc.v = wmma_bf16(fa.v, fb.v, acc.v);
      }
      int colBase = nT * 16;
      float qb = qkv_b[colBase + lr];
      if (nT < 16) {
        // q (cols 0..127) and k (cols 128..255): natural row-major
        for (int j = 0; j < 8; ++j) {
          int m = mT * 16 + j + 8 * lh;
          sQKV[m * 384 + colBase + lr] = (__bf16)(acc.f[j] + qb);
        }
      } else {
        // v section: store transposed vT(h, d, m) so P@v B-frags are contiguous
        int h = (nT - 16) >> 1;
        int d = ((nT - 16) & 1) * 16 + lr;
        int vbase = (h * 16 + (d >> 1)) * 384 + 256 + (d & 1) * 64;
        for (int j = 0; j < 8; ++j) {
          int m = mT * 16 + j + 8 * lh;
          sQKV[vbase + m] = (__bf16)(acc.f[j] + qb);
        }
      }
    }
  }
  __syncthreads();

  float*  sS = sScr;
  __bf16* sP = (__bf16*)sScr;     // P row r at byte r*256 (stride 128 elems)
  const float* maskw = mask + (size_t)(b & 63) * NSEQ * NSEQ;

  for (int h = 0; h < NHEAD; ++h) {
    // ---- stage 2: S = q_h @ k_h^T  (64x64, K=32 -> 1 WMMA/tile, 2/wave) ----
    for (int i = 0; i < 2; ++i) {
      int t = wv * 2 + i;
      int mT = t >> 2, nT = t & 3;
      FragB fa, fb;
      FragC acc; acc.v = (f32x8){0.f,0.f,0.f,0.f,0.f,0.f,0.f,0.f};
      const __bf16* arow = sQKV + (mT * 16 + lr) * 384 + h * HD + 16 * lh;
      fa.q[0] = *(const uint4*)(arow);
      fa.q[1] = *(const uint4*)(arow + 8);
      // B[kk][n] = k_h[n][kk] : 16 contiguous kk per lane
      const __bf16* brow = sQKV + (nT * 16 + lr) * 384 + CDIM + h * HD + 16 * lh;
      fb.q[0] = *(const uint4*)(brow);
      fb.q[1] = *(const uint4*)(brow + 8);
      acc.v = wmma_bf16(fa.v, fb.v, acc.v);
      for (int j = 0; j < 8; ++j) {
        int m = mT * 16 + j + 8 * lh;
        sS[m * 64 + nT * 16 + lr] = acc.f[j];
      }
    }
    __syncthreads();

    // ---- softmax: 4 lanes per row (same wave), register-resident segment ----
    {
      int r  = tid >> 2;
      int c0 = (tid & 3) * 16;
      uint4* dst = (uint4*)(sP + r * 128 + c0);
      if (r < NSEQ) {
        const float* srow = sS + r * 64;
        const float* bh = bias + h * (NSEQ * NSEQ) + r * NSEQ;
        const float* mr = maskw + r * NSEQ;
        float sv[16];
        float mx = -1e30f;
#pragma unroll
        for (int i = 0; i < 16; ++i) {
          int c = c0 + i;
          float s = (c < NSEQ) ? (srow[c] * SCALE + bh[c] + mr[c]) : -1e30f;
          sv[i] = s;
          mx = fmaxf(mx, s);
        }
        mx = fmaxf(mx, __shfl_xor(mx, 1, 32));
        mx = fmaxf(mx, __shfl_xor(mx, 2, 32));
        float sum = 0.f;
#pragma unroll
        for (int i = 0; i < 16; ++i) {
          float e = __expf(sv[i] - mx);   // c>=NSEQ underflows to 0
          sv[i] = e;
          sum += e;
        }
        sum += __shfl_xor(sum, 1, 32);
        sum += __shfl_xor(sum, 2, 32);
        float inv = 1.0f / sum;
        union { uint4 q[2]; __bf16 hx[16]; } pk;
#pragma unroll
        for (int i = 0; i < 16; ++i) {
          int c = c0 + i;
          pk.hx[i] = (__bf16)((c < NSEQ) ? sv[i] * inv : 0.0f);
        }
        // all LDS reads above precede these writes in wave lockstep: safe alias
        dst[0] = pk.q[0];
        dst[1] = pk.q[1];
      } else {
        uint4 z = {0u, 0u, 0u, 0u};
        dst[0] = z;
        dst[1] = z;
      }
    }
    __syncthreads();

    // ---- stage 3: O_h = P(64x64) @ v_h(64x32) -> overwrite dead q_h slice --
    {
      int mT = wv >> 1, nT = wv & 1;
      FragC acc; acc.v = (f32x8){0.f,0.f,0.f,0.f,0.f,0.f,0.f,0.f};
      for (int ks = 0; ks < 2; ++ks) {
        FragB fa, fb;
        const __bf16* arow = sP + (mT * 16 + lr) * 128 + ks * 32 + 16 * lh;
        fa.q[0] = *(const uint4*)(arow);
        fa.q[1] = *(const uint4*)(arow + 8);
        int d = nT * 16 + lr;
        const __bf16* brow = sQKV + (h * 16 + (d >> 1)) * 384 + 256 +
                             (d & 1) * 64 + ks * 32 + 16 * lh;
        fb.q[0] = *(const uint4*)(brow);
        fb.q[1] = *(const uint4*)(brow + 8);
        acc.v = wmma_bf16(fa.v, fb.v, acc.v);
      }
      for (int j = 0; j < 8; ++j) {
        int m = mT * 16 + j + 8 * lh;
        sQKV[m * 384 + h * HD + nT * 16 + lr] = (__bf16)acc.f[j];
      }
    }
    __syncthreads();
  }

  // ---- stage 4: out = O(64x128) @ Wproj^T(128x128) + proj_b  (4 tiles/wave)
  {
    float* ob = out + (size_t)b * NSEQ * CDIM;
    for (int i = 0; i < 4; ++i) {
      int t = wv * 4 + i;
      int mT = t & 3, nT = t >> 2;
      FragC acc; acc.v = (f32x8){0.f,0.f,0.f,0.f,0.f,0.f,0.f,0.f};
      for (int ks = 0; ks < 4; ++ks) {
        FragB fa, fb;
        const __bf16* arow = sQKV + (mT * 16 + lr) * 384 + ks * 32 + 16 * lh;
        fa.q[0] = *(const uint4*)(arow);
        fa.q[1] = *(const uint4*)(arow + 8);
        const uint4* bp = (const uint4*)(wp + (nT * 4 + ks) * 512 + lane * 16);
        fb.q[0] = bp[0];
        fb.q[1] = bp[1];
        acc.v = wmma_bf16(fa.v, fb.v, acc.v);
      }
      int col = nT * 16 + lr;
      float pb = proj_b[col];
      for (int j = 0; j < 8; ++j) {
        int m = mT * 16 + j + 8 * lh;
        if (m < NSEQ) ob[m * CDIM + col] = acc.f[j] + pb;
      }
    }
  }
}

extern "C" void kernel_launch(void* const* d_in, const int* in_sizes, int n_in,
                              void* d_out, int out_size, void* d_ws, size_t ws_size,
                              hipStream_t stream) {
  (void)in_sizes; (void)n_in; (void)out_size; (void)ws_size;
  const float* x       = (const float*)d_in[0];
  const float* mask    = (const float*)d_in[1];
  const float* qkv_w   = (const float*)d_in[2];
  const float* qkv_b   = (const float*)d_in[3];
  const float* rpb     = (const float*)d_in[4];
  const float* proj_w  = (const float*)d_in[5];
  const float* proj_b  = (const float*)d_in[6];
  const int*   rel_idx = (const int*)d_in[7];

  __bf16* wq   = (__bf16*)((char*)d_ws + WS_QKV_OFF);
  __bf16* wp   = (__bf16*)((char*)d_ws + WS_PROJ_OFF);
  float*  bias = (float*)((char*)d_ws + WS_BIAS_OFF);

  // prep: 32768 weight dwords + 9604 bias elems
  int prep_threads = 32768 + NHEAD * NSEQ * NSEQ;
  int prep_blocks = (prep_threads + 255) / 256;
  swin_prep<<<prep_blocks, 256, 0, stream>>>(qkv_w, proj_w, rpb, rel_idx,
                                             wq, wp, bias);

  swin_attn<<<4096, 256, 0, stream>>>(x, mask, qkv_b, proj_b,
                                      wq, wp, bias, (float*)d_out);
}
